// OneTransformLayer_26688926778134
// MI455X (gfx1250) — matrix operand order repaired and checked by reading, not compiled, not run on hardware
//
#include <hip/hip_runtime.h>

// ---------------- types ----------------
typedef unsigned short bf16_t;                                   // raw bf16 bits
typedef __bf16 v16bf __attribute__((ext_vector_type(16)));
typedef float  v8f   __attribute__((ext_vector_type(8)));

union FragU { uint4 q[2]; v16bf v; };
union F8    { v8f v; float f[8]; };

#define Bsz  8
#define Ssz  512
#define Hsz  1024
#define NHsz 16
#define DHsz 64
#define FFsz 4096
#define Msz  (Bsz * Ssz)   // 4096 rows

// float -> bf16 (round to nearest even)
__device__ __forceinline__ bf16_t f2bf(float f) {
  unsigned int u = __float_as_uint(f);
  u += 0x7FFFu + ((u >> 16) & 1u);
  return (bf16_t)(u >> 16);
}

// WMMA: D = A(16x32 bf16) * B(32x16 bf16) + C(16x16 f32)
__device__ __forceinline__ v8f wmma_bf16(v16bf a, v16bf b, v8f c) {
  return __builtin_amdgcn_wmma_f32_16x16x32_bf16(false, a, false, b, (short)0, c,
                                                 false, false);
}

// A-matrix fragment (16x32, 16-bit): lane L holds row M = L%16.
// lanes 0-15: K = {0..7, 16..23}; lanes 16-31: K = {8..15, 24..31}
__device__ __forceinline__ v16bf load_frag_a(const bf16_t* tile, int ld, int lane) {
  int r = lane & 15, half = (lane >> 4) & 1;
  const bf16_t* p = tile + (size_t)r * ld + half * 8;
  FragU f;
  f.q[0] = *(const uint4*)(p);
  f.q[1] = *(const uint4*)(p + 16);
  return f.v;
}

// B-matrix fragment (32x16, 16-bit), source Bt[N][K] row-major:
// lane L holds column N = L%16; lanes 0-15: K = {0..7, 16..23}; lanes 16-31: K = {8..15, 24..31}
__device__ __forceinline__ v16bf load_frag_b(const bf16_t* tile, int ld, int lane) {
  int n = lane & 15, half = (lane >> 4) & 1;
  const bf16_t* p = tile + (size_t)n * ld + half * 8;
  FragU f;
  f.q[0] = *(const uint4*)(p);
  f.q[1] = *(const uint4*)(p + 16);
  return f.v;
}

// CDNA5 async global->LDS copy (per-lane 16B), tracked by ASYNCcnt.
__device__ __forceinline__ void async_copy_b128(const bf16_t* gsrc, bf16_t* ldst) {
  unsigned lofs = (unsigned)(uintptr_t)ldst;   // LDS offset = low 32 bits of generic addr
  asm volatile("global_load_async_to_lds_b128 %0, %1, off"
               :: "v"(lofs), "v"(gsrc) : "memory");
}
__device__ __forceinline__ void wait_async0() {
  asm volatile("s_wait_asynccnt 0x0" ::: "memory");
}

__device__ __forceinline__ float gelu_t(float x) {
  float x3 = x * x * x;
  return 0.5f * x * (1.0f + tanhf(0.7978845608028654f * (x + 0.044715f * x3)));
}

// ---------------- elementwise f32 -> bf16 ----------------
__global__ __launch_bounds__(256) void cvt_bf16_kernel(const float* __restrict__ in,
                                                       bf16_t* __restrict__ out, int n) {
  int i = blockIdx.x * 256 + threadIdx.x;
  if (i < n) out[i] = f2bf(in[i]);
}

// ---------------- transpose + convert: W[K][N] f32 -> Wt[N][K] bf16 ----------------
__global__ __launch_bounds__(256) void transpose_cvt_kernel(const float* __restrict__ W,
                                                            bf16_t* __restrict__ Wt,
                                                            int K, int N) {
  __shared__ float t[32][33];
  int n0 = blockIdx.x * 32, k0 = blockIdx.y * 32;
  int tx = threadIdx.x, ty = threadIdx.y;
  #pragma unroll
  for (int j = ty; j < 32; j += 8)
    t[j][tx] = W[(size_t)(k0 + j) * N + n0 + tx];
  __syncthreads();
  #pragma unroll
  for (int j = ty; j < 32; j += 8)
    Wt[(size_t)(n0 + j) * K + k0 + tx] = f2bf(t[tx][j]);
}

// ---------------- GEMM: C[M,N] = alpha * (A[M,K] * Bt[N,K]^T + bias) ----------------
// BK=64 double-buffered LDS staging via async global->LDS (ASYNCcnt pipeline):
// 16 WMMAs per wave between barriers, next tile's copy overlapped with compute.
// MODE 0: f32 out   1: bf16 out   2: gelu->bf16 out   3: head-transpose bf16 ([b,h,d,s])
template <int MODE>
__global__ __launch_bounds__(256) void gemm_bf16_kernel(
    const bf16_t* __restrict__ A, int lda, const bf16_t* __restrict__ Bt, int ldb,
    const float* __restrict__ bias, float alpha, int K,
    float* __restrict__ Cf, bf16_t* __restrict__ Cb, int ldc) {
  __shared__ __align__(16) bf16_t sA[2][128 * 64];
  __shared__ __align__(16) bf16_t sB[2][128 * 64];

  int tid = threadIdx.x, lane = tid & 31, wid = tid >> 5;
  int m0 = blockIdx.y * 128, n0 = blockIdx.x * 128;
  int wm = (wid & 1) * 64;      // 2 wave rows
  int wn = (wid >> 1) * 32;     // 4 wave cols

  v8f acc[4][2];
  #pragma unroll
  for (int mt = 0; mt < 4; ++mt)
    #pragma unroll
    for (int nt = 0; nt < 2; ++nt) acc[mt][nt] = (v8f)0.0f;

  // issue one 128x64 A tile + 128x64 Bt tile into buffer ib (8 async ops / wave)
  auto issue = [&](int ib, int k0) {
    #pragma unroll
    for (int c = tid; c < 1024; c += 256) {
      int r = c >> 3, col = (c & 7) << 3;
      async_copy_b128(A + (size_t)(m0 + r) * lda + k0 + col, &sA[ib][r * 64 + col]);
      async_copy_b128(Bt + (size_t)(n0 + r) * ldb + k0 + col, &sB[ib][r * 64 + col]);
    }
  };

  issue(0, 0);
  for (int k0 = 0; k0 < K; k0 += 64) {
    int ib = (k0 >> 6) & 1;
    wait_async0();        // this wave's writes into buf ib have landed in LDS
    __syncthreads();      // all waves' writes visible; all done reading buf ib^1
    if (k0 + 64 < K) issue(ib ^ 1, k0 + 64);   // overlap next copy with compute
    #pragma unroll
    for (int ks = 0; ks < 2; ++ks) {
      v16bf bfr0 = load_frag_b(sB[ib] + (size_t)(wn + 0) * 64 + ks * 32, 64, lane);
      v16bf bfr1 = load_frag_b(sB[ib] + (size_t)(wn + 16) * 64 + ks * 32, 64, lane);
      #pragma unroll
      for (int mt = 0; mt < 4; ++mt) {
        v16bf afr = load_frag_a(sA[ib] + (size_t)(wm + mt * 16) * 64 + ks * 32, 64, lane);
        acc[mt][0] = wmma_bf16(afr, bfr0, acc[mt][0]);
        acc[mt][1] = wmma_bf16(afr, bfr1, acc[mt][1]);
      }
    }
  }

  int cn = lane & 15, chalf = lane >> 4;
  #pragma unroll
  for (int nt = 0; nt < 2; ++nt) {
    int n = n0 + wn + nt * 16 + cn;
    float bv = bias ? bias[n] : 0.0f;
    #pragma unroll
    for (int mt = 0; mt < 4; ++mt) {
      F8 u; u.v = acc[mt][nt];
      #pragma unroll
      for (int e = 0; e < 8; ++e) {
        int m = m0 + wm + mt * 16 + e + 8 * chalf;
        float val = alpha * (u.f[e] + bv);
        if (MODE == 0) {
          Cf[(size_t)m * ldc + n] = val;
        } else if (MODE == 1) {
          Cb[(size_t)m * ldc + n] = f2bf(val);
        } else if (MODE == 2) {
          Cb[(size_t)m * ldc + n] = f2bf(gelu_t(val));
        } else {  // [b*S+s, h*64+d] -> [(b*16+h)*64+d][s]
          int bb = m >> 9, s = m & 511, hh = n >> 6, d = n & 63;
          Cb[((size_t)((bb << 4) + hh) * 64 + d) * 512 + s] = f2bf(val);
        }
      }
    }
  }
}

// ---------------- flash attention ----------------
// Q [4096,1024] bf16 (pre-scaled by 1/8), K [4096,1024] bf16, Vt [128][64][512] bf16
// ctx [4096,1024] bf16. Block = 4 waves, one (b,h) x 64 query rows.
__global__ __launch_bounds__(128) void attn_kernel(
    const bf16_t* __restrict__ Q, const bf16_t* __restrict__ Kb,
    const bf16_t* __restrict__ Vt, const int* __restrict__ mask,
    bf16_t* __restrict__ ctx) {
  int head = blockIdx.x;             // b*16 + h
  int b = head >> 4, h = head & 15;
  int tid = threadIdx.x, w = tid >> 5, lane = tid & 31;
  int cn = lane & 15, chalf = lane >> 4;
  int q0 = blockIdx.y * 64 + w * 16; // wave's first query row within S

  const bf16_t* Qh = Q + (size_t)(b * Ssz) * Hsz + h * DHsz;
  const bf16_t* Kh = Kb + (size_t)(b * Ssz) * Hsz + h * DHsz;
  const bf16_t* Vh = Vt + (size_t)head * DHsz * Ssz;      // [64][512]
  const int* mk = mask + (size_t)b * Ssz * Ssz;

  __shared__ __align__(16) bf16_t sP[4][16 * 64];
  bf16_t* pw = sP[w];

  v16bf aQ0 = load_frag_a(Qh + (size_t)q0 * Hsz + 0,  Hsz, lane);
  v16bf aQ1 = load_frag_a(Qh + (size_t)q0 * Hsz + 32, Hsz, lane);

  v8f acc[4];
  #pragma unroll
  for (int dt = 0; dt < 4; ++dt) acc[dt] = (v8f)0.0f;
  float ms[8], ls[8];
  #pragma unroll
  for (int e = 0; e < 8; ++e) { ms[e] = -3.0e38f; ls[e] = 0.0f; }

  for (int j = 0; j < 8; ++j) {      // key chunks of 64
    int k0 = j * 64;
    v8f sc[4];
    #pragma unroll
    for (int nt = 0; nt < 4; ++nt) sc[nt] = (v8f)0.0f;
    #pragma unroll
    for (int nt = 0; nt < 4; ++nt) {
      v16bf bK0 = load_frag_b(Kh + (size_t)(k0 + nt * 16) * Hsz + 0,  Hsz, lane);
      sc[nt] = wmma_bf16(aQ0, bK0, sc[nt]);
      v16bf bK1 = load_frag_b(Kh + (size_t)(k0 + nt * 16) * Hsz + 32, Hsz, lane);
      sc[nt] = wmma_bf16(aQ1, bK1, sc[nt]);
    }
    // prefetch next chunk's K / V rows while softmax VALU runs
    if (j < 7) {
      #pragma unroll
      for (int nt = 0; nt < 4; ++nt)
        __builtin_prefetch(Kh + (size_t)(k0 + 64 + nt * 16 + cn) * Hsz, 0, 0);
      #pragma unroll
      for (int dt = 0; dt < 4; ++dt)
        __builtin_prefetch(Vh + (size_t)(dt * 16 + cn) * Ssz + k0 + 64, 0, 0);
    }
    // mask + online softmax (rows live across the 16 lanes of each half)
    F8 s4[4];
    #pragma unroll
    for (int nt = 0; nt < 4; ++nt) s4[nt].v = sc[nt];
    float scl[8];
    #pragma unroll
    for (int e = 0; e < 8; ++e) {
      int q = q0 + e + 8 * chalf;
      float mx = -3.0e38f;
      #pragma unroll
      for (int nt = 0; nt < 4; ++nt) {
        int kk = k0 + nt * 16 + cn;
        s4[nt].f[e] += (1.0f - (float)mk[(size_t)q * Ssz + kk]) * -10000.0f;
        mx = fmaxf(mx, s4[nt].f[e]);
      }
      #pragma unroll
      for (int o = 8; o >= 1; o >>= 1) mx = fmaxf(mx, __shfl_xor(mx, o, 32));
      float mnew = fmaxf(ms[e], mx);
      scl[e] = __expf(ms[e] - mnew);
      float rs = 0.0f;
      #pragma unroll
      for (int nt = 0; nt < 4; ++nt) {
        float p = __expf(s4[nt].f[e] - mnew);
        s4[nt].f[e] = p;
        rs += p;
      }
      #pragma unroll
      for (int o = 8; o >= 1; o >>= 1) rs += __shfl_xor(rs, o, 32);
      ls[e] = ls[e] * scl[e] + rs;
      ms[e] = mnew;
    }
    #pragma unroll
    for (int dt = 0; dt < 4; ++dt) {
      F8 u; u.v = acc[dt];
      #pragma unroll
      for (int e = 0; e < 8; ++e) u.f[e] *= scl[e];
      acc[dt] = u.v;
    }
    // P (C-layout) -> LDS -> A-layout fragments (per-wave region; LDS in-order per wave)
    #pragma unroll
    for (int e = 0; e < 8; ++e) {
      int prow = e + 8 * chalf;
      #pragma unroll
      for (int nt = 0; nt < 4; ++nt)
        pw[prow * 64 + nt * 16 + cn] = f2bf(s4[nt].f[e]);
    }
    __asm__ volatile("" ::: "memory");
    #pragma unroll
    for (int ks = 0; ks < 2; ++ks) {
      v16bf aP = load_frag_a(pw + ks * 32, 64, lane);
      #pragma unroll
      for (int dt = 0; dt < 4; ++dt) {
        v16bf bV = load_frag_b(Vh + (size_t)(dt * 16) * Ssz + k0 + ks * 32, Ssz, lane);
        acc[dt] = wmma_bf16(aP, bV, acc[dt]);
      }
    }
  }
  // normalize + store ctx bf16
  #pragma unroll
  for (int e = 0; e < 8; ++e) {
    int q = q0 + e + 8 * chalf;
    float inv = 1.0f / ls[e];
    #pragma unroll
    for (int dt = 0; dt < 4; ++dt) {
      F8 u; u.v = acc[dt];
      ctx[(size_t)(b * Ssz + q) * Hsz + h * DHsz + dt * 16 + cn] = f2bf(u.f[e] * inv);
    }
  }
}

// ---------------- residual + layernorm ----------------
__device__ __forceinline__ float block_sum(float v, float* red, int tid) {
  #pragma unroll
  for (int o = 16; o >= 1; o >>= 1) v += __shfl_xor(v, o, 32);
  int w = tid >> 5, lane = tid & 31;
  __syncthreads();
  if (lane == 0) red[w] = v;
  __syncthreads();
  float x = red[0];
  #pragma unroll
  for (int i = 1; i < 8; ++i) x += red[i];
  return x;
}

__global__ __launch_bounds__(256) void ln_kernel(
    const float* __restrict__ proj, const float* __restrict__ resid,
    const float* __restrict__ gamma, const float* __restrict__ beta,
    float* __restrict__ outf, bf16_t* __restrict__ outb) {
  const int H = Hsz;
  int row = blockIdx.x, tid = threadIdx.x;
  __shared__ float xs[Hsz];
  __shared__ float red[8];
  float s = 0.0f;
  for (int i = tid; i < H; i += 256) {
    float v = proj[(size_t)row * H + i] + resid[(size_t)row * H + i];
    xs[i] = v;
    s += v;
  }
  float mean = block_sum(s, red, tid) * (1.0f / H);
  float vs = 0.0f;
  for (int i = tid; i < H; i += 256) {
    float d = xs[i] - mean;
    vs += d * d;
  }
  float var = block_sum(vs, red, tid) * (1.0f / H);
  float rstd = rsqrtf(var + 1e-12f);
  for (int i = tid; i < H; i += 256) {
    float y = (xs[i] - mean) * rstd * gamma[i] + beta[i];
    outf[(size_t)row * H + i] = y;
    if (outb) outb[(size_t)row * H + i] = f2bf(y);
  }
}

// ---------------- launch ----------------
extern "C" void kernel_launch(void* const* d_in, const int* in_sizes, int n_in,
                              void* d_out, int out_size, void* d_ws, size_t ws_size,
                              hipStream_t stream) {
  const float* X    = (const float*)d_in[0];
  const int*   mask = (const int*)d_in[1];
  const float* Wq = (const float*)d_in[2];  const float* bq = (const float*)d_in[3];
  const float* Wk = (const float*)d_in[4];  const float* bk = (const float*)d_in[5];
  const float* Wv = (const float*)d_in[6];  const float* bv = (const float*)d_in[7];
  const float* Wao = (const float*)d_in[8]; const float* bao = (const float*)d_in[9];
  const float* g1 = (const float*)d_in[10]; const float* be1 = (const float*)d_in[11];
  const float* Wi = (const float*)d_in[12]; const float* bi = (const float*)d_in[13];
  const float* Wo = (const float*)d_in[14]; const float* bo = (const float*)d_in[15];
  const float* g2 = (const float*)d_in[16]; const float* be2 = (const float*)d_in[17];
  float* out = (float*)d_out;

  char* cur = (char*)d_ws;
  auto take = [&](size_t bytes) {
    char* p = cur;
    cur += (bytes + 255) & ~(size_t)255;
    return p;
  };
  bf16_t* Xbf  = (bf16_t*)take((size_t)Msz * Hsz * 2);
  bf16_t* WqT  = (bf16_t*)take((size_t)Hsz * Hsz * 2);
  bf16_t* WkT  = (bf16_t*)take((size_t)Hsz * Hsz * 2);
  bf16_t* WvT  = (bf16_t*)take((size_t)Hsz * Hsz * 2);
  bf16_t* WaoT = (bf16_t*)take((size_t)Hsz * Hsz * 2);
  bf16_t* WiT  = (bf16_t*)take((size_t)FFsz * Hsz * 2);   // Wi^T [FF,H]
  bf16_t* WoT  = (bf16_t*)take((size_t)Hsz * FFsz * 2);   // Wo^T [H,FF]
  // Q,K,Vt,ctx are contiguous (4 x 8MB); inter (32MB) aliases them after attention.
  bf16_t* Qbf  = (bf16_t*)take((size_t)Msz * Hsz * 2);
  bf16_t* Kbf  = (bf16_t*)take((size_t)Msz * Hsz * 2);
  bf16_t* Vtb  = (bf16_t*)take((size_t)Msz * Hsz * 2);
  bf16_t* ctxb = (bf16_t*)take((size_t)Msz * Hsz * 2);
  float*  attn_proj  = (float*)take((size_t)Msz * Hsz * 4);  // also reused as ffn_proj
  float*  attn_out_f = (float*)take((size_t)Msz * Hsz * 4);
  bf16_t* attn_out_b = (bf16_t*)take((size_t)Msz * Hsz * 2);
  bf16_t* interb   = Qbf;        // 32MB alias over Q/K/Vt/ctx
  float*  ffn_proj = attn_proj;  // alias

  // 1. conversions
  cvt_bf16_kernel<<<(Msz * Hsz) / 256, 256, 0, stream>>>(X, Xbf, Msz * Hsz);
  dim3 tb(32, 8);
  transpose_cvt_kernel<<<dim3(Hsz / 32, Hsz / 32), tb, 0, stream>>>(Wq, WqT, Hsz, Hsz);
  transpose_cvt_kernel<<<dim3(Hsz / 32, Hsz / 32), tb, 0, stream>>>(Wk, WkT, Hsz, Hsz);
  transpose_cvt_kernel<<<dim3(Hsz / 32, Hsz / 32), tb, 0, stream>>>(Wv, WvT, Hsz, Hsz);
  transpose_cvt_kernel<<<dim3(Hsz / 32, Hsz / 32), tb, 0, stream>>>(Wao, WaoT, Hsz, Hsz);
  transpose_cvt_kernel<<<dim3(FFsz / 32, Hsz / 32), tb, 0, stream>>>(Wi, WiT, Hsz, FFsz);
  transpose_cvt_kernel<<<dim3(Hsz / 32, FFsz / 32), tb, 0, stream>>>(Wo, WoT, FFsz, Hsz);

  // 2. QKV projections (Q pre-scaled by 1/sqrt(DH); V stored head-transposed)
  dim3 gHH(Hsz / 128, Msz / 128);
  gemm_bf16_kernel<1><<<gHH, 256, 0, stream>>>(Xbf, Hsz, WqT, Hsz, bq, 0.125f, Hsz,
                                               nullptr, Qbf, Hsz);
  gemm_bf16_kernel<1><<<gHH, 256, 0, stream>>>(Xbf, Hsz, WkT, Hsz, bk, 1.0f, Hsz,
                                               nullptr, Kbf, Hsz);
  gemm_bf16_kernel<3><<<gHH, 256, 0, stream>>>(Xbf, Hsz, WvT, Hsz, bv, 1.0f, Hsz,
                                               nullptr, Vtb, 0);

  // 3. attention
  attn_kernel<<<dim3(Bsz * NHsz, Ssz / 64), 128, 0, stream>>>(Qbf, Kbf, Vtb, mask, ctxb);

  // 4. output projection + residual layernorm
  gemm_bf16_kernel<0><<<gHH, 256, 0, stream>>>(ctxb, Hsz, WaoT, Hsz, bao, 1.0f, Hsz,
                                               attn_proj, nullptr, Hsz);
  ln_kernel<<<Msz, 256, 0, stream>>>(attn_proj, X, g1, be1, attn_out_f, attn_out_b);

  // 5. FFN
  gemm_bf16_kernel<2><<<dim3(FFsz / 128, Msz / 128), 256, 0, stream>>>(
      attn_out_b, Hsz, WiT, Hsz, bi, 1.0f, Hsz, nullptr, interb, FFsz);
  gemm_bf16_kernel<0><<<gHH, 256, 0, stream>>>(interb, FFsz, WoT, FFsz, bo, 1.0f, FFsz,
                                               ffn_proj, nullptr, Hsz);
  ln_kernel<<<Msz, 256, 0, stream>>>(ffn_proj, attn_out_f, g2, be2, out, nullptr);
}
